// Decoder_78752520339746
// MI455X (gfx1250) — matrix operand order repaired and checked by reading, not compiled, not run on hardware
//
#include <hip/hip_runtime.h>
#include <hip/hip_bf16.h>

// Problem constants (fixed by the reference).
#define TQ  64
#define TKK 96
#define BB  32
#define HH  512
#define H4  2048
#define TQB (TQ * BB)   // 2048 query rows
#define TKB (TKK * BB)  // 3072 encoder rows
#define NEGV (-1000000.0f)

typedef __attribute__((ext_vector_type(16))) __bf16       v16bf;
typedef __attribute__((ext_vector_type(8)))  float        v8f;
typedef __attribute__((ext_vector_type(4)))  unsigned int v4u;

union FragAB { v16bf h; v4u q[2]; };

__device__ __forceinline__ unsigned short f2bf(float x) {
  unsigned int u = __float_as_uint(x);
  u += 0x7fffu + ((u >> 16) & 1u);   // round-to-nearest-even
  return (unsigned short)(u >> 16);
}
__device__ __forceinline__ float sigm(float x) { return 1.0f / (1.0f + __expf(-x)); }

__device__ __forceinline__ float waveSum(float v) {
  #pragma unroll
  for (int o = 16; o > 0; o >>= 1) v += __shfl_xor(v, o, 32);
  return v;
}
__device__ __forceinline__ float waveMax(float v) {
  #pragma unroll
  for (int o = 16; o > 0; o >>= 1) v = fmaxf(v, __shfl_xor(v, o, 32));
  return v;
}

// Fragment loads per CDNA5 ISA section 7.12.2 layouts.
// A (16x32 bf16, MxK): lane = m | (s<<4); elems 0..7 -> K = k0+8s..+7,
//                      elems 8..15 -> K = k0+16+8s..+7  (two 16B chunks)
__device__ __forceinline__ void loadA(FragAB& f, const unsigned short* arow,
                                      int k0, int s) {
  f.q[0] = *reinterpret_cast<const v4u*>(arow + k0 + s * 8);
  f.q[1] = *reinterpret_cast<const v4u*>(arow + k0 + 16 + s * 8);
}
// B (32x16 bf16, KxN) from row-major W[N,K]: lane = n | (s<<4);
// elems 0..7 -> K = k0+8s..+7, elems 8..15 -> K = k0+16+8s..+7 (two 16B chunks of row n)
__device__ __forceinline__ void loadB(FragAB& f, const unsigned short* wrow,
                                      int k0, int s) {
  f.q[0] = *reinterpret_cast<const v4u*>(wrow + k0 + s * 8);
  f.q[1] = *reinterpret_cast<const v4u*>(wrow + k0 + 16 + s * 8);
}
__device__ __forceinline__ v8f wmma_bf16(const FragAB& a, const FragAB& b, v8f c) {
  return __builtin_amdgcn_wmma_f32_16x16x32_bf16(false, a.h, false, b.h,
                                                 (short)0, c, false, false);
}

// ---------------- f32 -> bf16 conversion ----------------
__global__ void cvt_f32_bf16(const float* __restrict__ src,
                             unsigned short* __restrict__ dst, int n) {
  for (int i = blockIdx.x * blockDim.x + threadIdx.x; i < n;
       i += gridDim.x * blockDim.x)
    dst[i] = f2bf(src[i]);
}

// ------- generic C[M,N] = A[M,K] * W[N,K]^T + bias[N] -------
// One 16x32 output slab (2 N-tiles) per wave; software-pipelined (ping-pong)
// fragment loads so v_wmma overlaps the next fragment fetch.
// Requires K % 64 == 0 (true here: 512 / 1024).
__global__ __launch_bounds__(128)
void gemm_bias_wmma(const unsigned short* __restrict__ A,
                    const unsigned short* __restrict__ W,
                    const float* __restrict__ bias,
                    float* __restrict__ C, int M, int N, int K) {
  const int lane = threadIdx.x;
  const int tid  = blockIdx.x * 4 + threadIdx.y;
  const int ntp  = N >> 5;                  // 32-wide tile pairs along N
  const int numUnits = (M >> 4) * ntp;
  if (tid >= numUnits) return;
  const int mt = tid / ntp, np = tid - mt * ntp;
  const int m0 = mt << 4, n0 = np << 5;
  const int s = lane >> 4;
  const int m = m0 + (lane & 15), n = n0 + (lane & 15);

  v8f acc0, acc1;
  const float bv0 = bias ? bias[n] : 0.0f;
  const float bv1 = bias ? bias[n + 16] : 0.0f;
  #pragma unroll
  for (int r = 0; r < 8; ++r) { acc0[r] = bv0; acc1[r] = bv1; }

  const unsigned short* arow  = A + (size_t)m * K;
  const unsigned short* wrow0 = W + (size_t)n * K;
  const unsigned short* wrow1 = W + (size_t)(n + 16) * K;

  FragAB fa0, fa1, fb00, fb10, fb01, fb11;
  loadA(fa0, arow, 0, s);
  loadB(fb00, wrow0, 0, s);
  loadB(fb10, wrow1, 0, s);
  for (int k0 = 0; k0 < K; k0 += 64) {
    loadA(fa1, arow, k0 + 32, s);
    loadB(fb01, wrow0, k0 + 32, s);
    loadB(fb11, wrow1, k0 + 32, s);
    if (k0 + 128 < K) {
      __builtin_prefetch(wrow0 + k0 + 128, 0, 1);   // global_prefetch_b8
      __builtin_prefetch(wrow1 + k0 + 128, 0, 1);
    }
    acc0 = wmma_bf16(fa0, fb00, acc0);
    acc1 = wmma_bf16(fa0, fb10, acc1);
    if (k0 + 64 < K) {
      loadA(fa0, arow, k0 + 64, s);
      loadB(fb00, wrow0, k0 + 64, s);
      loadB(fb10, wrow1, k0 + 64, s);
    }
    acc0 = wmma_bf16(fa1, fb01, acc0);
    acc1 = wmma_bf16(fa1, fb11, acc1);
  }
  float* crow = C + (size_t)(m0 + 8 * s) * N + n;
  #pragma unroll
  for (int r = 0; r < 8; ++r) {
    crow[(size_t)r * N]      = acc0[r];
    crow[(size_t)r * N + 16] = acc1[r];
  }
}

// ------- recurrent step: z[32,2048] = xproj_t + h_bf16 @ Whh[2048,512]^T -------
// 128 wave-units of 16x32 output; same pipelined structure, K = 512 fixed.
__global__ __launch_bounds__(128)
void lstm_zstep_wmma(const unsigned short* __restrict__ hbf,
                     const unsigned short* __restrict__ Whh,
                     const float* __restrict__ xprojT,
                     float* __restrict__ z) {
  const int lane = threadIdx.x;
  const int tid  = blockIdx.x * 4 + threadIdx.y;   // 0..127 units (2 x 64)
  const int mt = tid >> 6, np = tid & 63;
  const int m0 = mt << 4, n0 = np << 5;
  const int s = lane >> 4;
  const int m = m0 + (lane & 15), n = n0 + (lane & 15);

  v8f acc0, acc1;
  const float* xp = xprojT + (size_t)(m0 + 8 * s) * H4 + n;
  #pragma unroll
  for (int r = 0; r < 8; ++r) {
    acc0[r] = xp[(size_t)r * H4];
    acc1[r] = xp[(size_t)r * H4 + 16];
  }

  const unsigned short* arow  = hbf + (size_t)m * HH;
  const unsigned short* wrow0 = Whh + (size_t)n * HH;
  const unsigned short* wrow1 = Whh + (size_t)(n + 16) * HH;

  FragAB fa0, fa1, fb00, fb10, fb01, fb11;
  loadA(fa0, arow, 0, s);
  loadB(fb00, wrow0, 0, s);
  loadB(fb10, wrow1, 0, s);
  #pragma unroll 2
  for (int k0 = 0; k0 < HH; k0 += 64) {
    loadA(fa1, arow, k0 + 32, s);
    loadB(fb01, wrow0, k0 + 32, s);
    loadB(fb11, wrow1, k0 + 32, s);
    acc0 = wmma_bf16(fa0, fb00, acc0);
    acc1 = wmma_bf16(fa0, fb10, acc1);
    if (k0 + 64 < HH) {
      loadA(fa0, arow, k0 + 64, s);
      loadB(fb00, wrow0, k0 + 64, s);
      loadB(fb10, wrow1, k0 + 64, s);
    }
    acc0 = wmma_bf16(fa1, fb01, acc0);
    acc1 = wmma_bf16(fa1, fb11, acc1);
  }
  float* zrow = z + (size_t)(m0 + 8 * s) * H4 + n;
  #pragma unroll
  for (int r = 0; r < 8; ++r) {
    zrow[(size_t)r * H4]      = acc0[r];
    zrow[(size_t)r * H4 + 16] = acc1[r];
  }
}

// ---------------- gates (fp32), writes y = h (+ residual), h in bf16 for next step ----
__global__ __launch_bounds__(256)
void lstm_gates(const float* __restrict__ z, float* __restrict__ c,
                const float* __restrict__ resid, float* __restrict__ y,
                unsigned short* __restrict__ hbf) {
  const int idx = blockIdx.x * 256 + threadIdx.x;   // 16384 = 32*512
  const int b = idx >> 9, j = idx & 511;
  const float* zr = z + (size_t)b * H4;
  const float zi = zr[j], zf = zr[512 + j], zg = zr[1024 + j], zo = zr[1536 + j];
  const float cn = sigm(zf) * c[idx] + sigm(zi) * tanhf(zg);
  const float h  = sigm(zo) * tanhf(cn);
  c[idx] = cn;
  hbf[idx] = f2bf(h);
  y[idx] = resid ? (h + resid[idx]) : h;
}

__global__ void init_hc(const float* __restrict__ h0, const float* __restrict__ c0,
                        unsigned short* __restrict__ hbf, float* __restrict__ c) {
  const int idx = blockIdx.x * 256 + threadIdx.x;   // 16384
  hbf[idx] = f2bf(h0[idx]);
  c[idx] = c0[idx];
}

// ---------------- fused additive attention: one wave per (q,b) ----------------
__global__ __launch_bounds__(256)
void attn_fused(const float* __restrict__ qp, const float* __restrict__ kp,
                const float* __restrict__ vp, const unsigned char* __restrict__ mask,
                const float* __restrict__ Ww, const float* __restrict__ WbArr,
                int wbIdx, float* __restrict__ ctx, int accum) {
  const int lane = threadIdx.x;
  const int wid  = blockIdx.x * 8 + threadIdx.y;   // 0..2047  == q*BB + b
  const int b = wid & 31;
  const float Wb = WbArr[wbIdx];

  float qv[16], wv[16];
  const float* qrow = qp + (size_t)wid * HH;
  #pragma unroll
  for (int j = 0; j < 16; ++j) { qv[j] = qrow[lane + 32 * j]; wv[j] = Ww[lane + 32 * j]; }

  float s0 = NEGV, s1 = NEGV, s2 = NEGV;
  for (int tk = 0; tk < TKK; ++tk) {
    const float* krow = kp + (size_t)(tk * BB + b) * HH;
    float p = 0.0f;
    #pragma unroll
    for (int j = 0; j < 16; ++j) p += tanhf(qv[j] + krow[lane + 32 * j]) * wv[j];
    float sc = waveSum(p) + Wb;                       // butterfly: all lanes get sum
    if (!mask[tk * BB + b]) sc = NEGV;
    if ((tk & 31) == lane) { if (tk < 32) s0 = sc; else if (tk < 64) s1 = sc; else s2 = sc; }
  }
  const float mx = waveMax(fmaxf(fmaxf(s0, s1), s2));
  const float e0 = __expf(s0 - mx), e1 = __expf(s1 - mx), e2 = __expf(s2 - mx);
  const float inv = 1.0f / waveSum(e0 + e1 + e2);

  float acc[16];
  #pragma unroll
  for (int j = 0; j < 16; ++j) acc[j] = 0.0f;
  for (int tk = 0; tk < TKK; ++tk) {
    const float e = (tk < 32) ? e0 : ((tk < 64) ? e1 : e2);
    const float wgt = __shfl(e, tk & 31, 32) * inv;
    const float* vrow = vp + (size_t)(tk * BB + b) * HH;
    #pragma unroll
    for (int j = 0; j < 16; ++j) acc[j] += wgt * vrow[lane + 32 * j];
  }
  float* crow = ctx + (size_t)wid * HH;
  #pragma unroll
  for (int j = 0; j < 16; ++j) {
    float v = acc[j];
    if (accum) v += crow[lane + 32 * j];
    crow[lane + 32 * j] = v;
  }
}

// ---------------- concat [tensor | c1+c2] -> bf16 [2048,1024] ----------------
__global__ void concat_to_bf16(const float* __restrict__ y2, const float* __restrict__ ctx,
                               unsigned short* __restrict__ out) {
  for (int i = blockIdx.x * blockDim.x + threadIdx.x; i < TQB * 1024;
       i += gridDim.x * blockDim.x) {
    const int r = i >> 10, jc = i & 1023;
    const float v = (jc < 512) ? y2[(size_t)r * 512 + jc]
                               : ctx[(size_t)r * 512 + (jc - 512)];
    out[i] = f2bf(v);
  }
}

// ======================= host orchestration =======================
extern "C" void kernel_launch(void* const* d_in, const int* in_sizes, int n_in,
                              void* d_out, int out_size, void* d_ws, size_t ws_size,
                              hipStream_t stream) {
  (void)in_sizes; (void)n_in; (void)out_size; (void)ws_size;
  const float* inputs = (const float*)d_in[0];
  const float* enc1   = (const float*)d_in[1];
  const float* enc2   = (const float*)d_in[2];
  const unsigned char* mask1 = (const unsigned char*)d_in[3];
  const unsigned char* mask2 = (const unsigned char*)d_in[4];
  const float* h0  = (const float*)d_in[5];
  const float* c0  = (const float*)d_in[6];
  const float* aQw = (const float*)d_in[7];
  const float* aQb = (const float*)d_in[8];
  const float* aKw = (const float*)d_in[9];
  const float* aKb = (const float*)d_in[10];
  const float* aVw = (const float*)d_in[11];
  const float* aVb = (const float*)d_in[12];
  const float* aWw = (const float*)d_in[13];
  const float* aWb = (const float*)d_in[14];
  const float* Wih = (const float*)d_in[15];
  const float* Whh = (const float*)d_in[16];
  const float* bb  = (const float*)d_in[17];
  const float* rWih = (const float*)d_in[18];
  const float* rWhh = (const float*)d_in[19];
  const float* rb   = (const float*)d_in[20];
  float* out = (float*)d_out;

  // Carve workspace (~75 MB), 256B aligned segments.
  char* ws = (char*)d_ws;
  size_t off = 0;
  auto carve = [&](size_t bytes) -> char* {
    char* p = ws + off;
    off = (off + bytes + 255) & ~(size_t)255;
    return p;
  };
  unsigned short* wih_bf  = (unsigned short*)carve((size_t)2 * H4 * HH * 2);
  unsigned short* whh_bf  = (unsigned short*)carve((size_t)2 * H4 * HH * 2);
  unsigned short* rwih_bf = (unsigned short*)carve((size_t)H4 * 1024 * 2);
  unsigned short* rwhh_bf = (unsigned short*)carve((size_t)H4 * HH * 2);
  unsigned short* qw_bf   = (unsigned short*)carve((size_t)2 * HH * HH * 2);
  unsigned short* kw_bf   = (unsigned short*)carve((size_t)2 * HH * HH * 2);
  unsigned short* vw_bf   = (unsigned short*)carve((size_t)2 * HH * HH * 2);
  unsigned short* x_bf    = (unsigned short*)carve((size_t)TQB * HH * 2);
  unsigned short* e1_bf   = (unsigned short*)carve((size_t)TKB * HH * 2);
  unsigned short* e2_bf   = (unsigned short*)carve((size_t)TKB * HH * 2);
  unsigned short* cat_bf  = (unsigned short*)carve((size_t)TQB * 1024 * 2);
  unsigned short* h_bf    = (unsigned short*)carve((size_t)BB * HH * 2);
  float* xproj = (float*)carve((size_t)TQB * H4 * 4);
  float* zbuf  = (float*)carve((size_t)BB * H4 * 4);
  float* cbuf  = (float*)carve((size_t)BB * HH * 4);
  float* y0    = (float*)carve((size_t)TQB * HH * 4);
  float* y1    = (float*)carve((size_t)TQB * HH * 4);
  float* qpb   = (float*)carve((size_t)TQB * HH * 4);
  float* kpb   = (float*)carve((size_t)TKB * HH * 4);
  float* vpb   = (float*)carve((size_t)TKB * HH * 4);
  float* ctxb  = (float*)carve((size_t)TQB * HH * 4);

  auto cvt = [&](const float* s, unsigned short* d, int n) {
    int g = (n + 255) / 256; if (g > 4096) g = 4096;
    cvt_f32_bf16<<<g, 256, 0, stream>>>(s, d, n);
  };
  auto gemm = [&](const unsigned short* A, const unsigned short* W,
                  const float* bias, float* C, int M, int N, int K) {
    int units = (M >> 4) * (N >> 5);
    gemm_bias_wmma<<<(units + 3) / 4, dim3(32, 4), 0, stream>>>(A, W, bias, C, M, N, K);
  };
  auto runLSTM = [&](const unsigned short* whhbf, const float* xp,
                     const float* h0l, const float* c0l,
                     const float* residBase, float* yBase) {
    init_hc<<<64, 256, 0, stream>>>(h0l, c0l, h_bf, cbuf);
    for (int t = 0; t < TQ; ++t) {
      lstm_zstep_wmma<<<32, dim3(32, 4), 0, stream>>>(
          h_bf, whhbf, xp + (size_t)t * BB * H4, zbuf);
      lstm_gates<<<64, 256, 0, stream>>>(
          zbuf, cbuf,
          residBase ? residBase + (size_t)t * BB * HH : (const float*)nullptr,
          yBase + (size_t)t * BB * HH, h_bf);
    }
  };

  // Weight conversions (bf16 operands, f32 accumulate).
  cvt(Wih,  wih_bf,  2 * H4 * HH);
  cvt(Whh,  whh_bf,  2 * H4 * HH);
  cvt(rWih, rwih_bf, H4 * 1024);
  cvt(rWhh, rwhh_bf, H4 * HH);
  cvt(aQw,  qw_bf,   2 * HH * HH);
  cvt(aKw,  kw_bf,   2 * HH * HH);
  cvt(aVw,  vw_bf,   2 * HH * HH);

  // Residual LSTM layer 0.
  cvt(inputs, x_bf, TQB * HH);
  gemm(x_bf, wih_bf, bb, xproj, TQB, H4, HH);
  runLSTM(whh_bf, xproj, h0, c0, inputs, y0);

  // Residual LSTM layer 1.
  cvt(y0, x_bf, TQB * HH);
  gemm(x_bf, wih_bf + (size_t)H4 * HH, bb + H4, xproj, TQB, H4, HH);
  runLSTM(whh_bf + (size_t)H4 * HH, xproj, h0 + BB * HH, c0 + BB * HH, y0, y1);

  // Two additive attentions over enc1/enc2; ctx accumulates c1+c2.
  cvt(y1,   x_bf,  TQB * HH);
  cvt(enc1, e1_bf, TKB * HH);
  cvt(enc2, e2_bf, TKB * HH);
  for (int i = 0; i < 2; ++i) {
    const unsigned short* ebf = (i == 0) ? e1_bf : e2_bf;
    const unsigned char*  msk = (i == 0) ? mask1 : mask2;
    gemm(x_bf, qw_bf + (size_t)i * HH * HH, aQb + i * HH, qpb, TQB, HH, HH);
    gemm(ebf,  kw_bf + (size_t)i * HH * HH, aKb + i * HH, kpb, TKB, HH, HH);
    gemm(ebf,  vw_bf + (size_t)i * HH * HH, aVb + i * HH, vpb, TKB, HH, HH);
    attn_fused<<<256, dim3(32, 8), 0, stream>>>(qpb, kpb, vpb, msk,
                                                aWw + i * HH, aWb, i, ctxb, i);
  }

  // Concat [tensor | c1+c2] and final LSTM (output straight to d_out).
  concat_to_bf16<<<2048, 256, 0, stream>>>(y1, ctxb, cat_bf);
  gemm(cat_bf, rwih_bf, rb, xproj, TQB, H4, 1024);
  runLSTM(rwhh_bf, xproj, h0 + 2 * BB * HH, c0 + 2 * BB * HH, nullptr, out);
}
